// WindowAttention_71390946394384
// MI455X (gfx1250) — matrix operand order repaired and checked by reading, not compiled, not run on hardware
//
#include <hip/hip_runtime.h>
#include <hip/hip_bf16.h>

typedef __bf16 bf16;
typedef __attribute__((ext_vector_type(16))) __bf16 v16bf;
typedef __attribute__((ext_vector_type(8)))  __bf16 v8bf;
typedef __attribute__((ext_vector_type(8)))  float  v8f;
typedef __attribute__((ext_vector_type(4)))  float  v4f;

#define NH_   12
#define DM_   768
#define HD_   64
#define L_    64
#define NWIN  512              // B(8) * 8 * 8 windows
#define TTOK  (NWIN * L_)      // 32768 tokens

// ---------------- WMMA fragment loaders (v_wmma_f32_16x16x32_bf16) ----------
// A fragment: 16(M) x 32(K), row-major source with leading dim ld.
// lane = m | (khalf<<4); elems j -> K = khalf*16 + j
__device__ __forceinline__ v16bf load_A_rm(const bf16* __restrict__ src, int ld,
                                           int row0, int col0, int lane) {
  const int m  = lane & 15;
  const int kh = lane >> 4;
  const bf16* p = src + (size_t)(row0 + m) * ld + col0 + kh * 16;
  return *(const v16bf*)p;            // K = col0 + kh*16 + [0..15]
}

// B fragment: 32(K) x 16(N) where B[k][n] = M[n0+n][k0+k], M row-major, ld elems.
// lane: N = lane&15, elems j -> K = (lane>>4)*16 + j  => one contiguous 32B load.
__device__ __forceinline__ v16bf load_B_nm(const bf16* __restrict__ M, int ld,
                                           int n0, int k0, int lane) {
  const bf16* p = M + (size_t)(n0 + (lane & 15)) * ld + k0 + (lane >> 4) * 16;
  return *(const v16bf*)p;
}

__device__ __forceinline__ v8f wmma_bf16(v16bf a, v16bf b, v8f c) {
  return __builtin_amdgcn_wmma_f32_16x16x32_bf16(false, a, false, b, (short)0, c,
                                                 false, false);
}

// ---------------- Stage 0: pack x (B,C,H,W) f32 -> window-major (T,768) bf16 --
// One block per (64-channel tile, image row h, batch b). Coalesced 64B reads,
// LDS transpose, 32B vector writes into token-major rows.
__global__ __launch_bounds__(256) void pack_x_kernel(const float* __restrict__ x,
                                                     bf16* __restrict__ Xw) {
  __shared__ bf16 tile[64][96] __attribute__((aligned(64)));   // [w][c], padded
  const int c0 = blockIdx.x * 64;
  const int h  = blockIdx.y;
  const int b  = blockIdx.z;
  {
    const int wseg = threadIdx.x & 3;         // which 16-float segment of w
    const int cc   = threadIdx.x >> 2;        // channel within tile
    const float* p = x + (((size_t)b * DM_ + c0 + cc) * 64 + h) * 64 + wseg * 16;
#pragma unroll
    for (int j = 0; j < 16; ++j)
      tile[wseg * 16 + j][cc] = (bf16)p[j];
  }
  __syncthreads();
  {
    const int cseg = threadIdx.x & 3;         // 16-channel segment
    const int wpix = threadIdx.x >> 2;        // w coordinate 0..63
    const int ww   = wpix >> 3;
    const int w_in = wpix & 7;
    const int t = ((b * 8 + (h >> 3)) * 8 + ww) * 64 + (h & 7) * 8 + w_in;
    v16bf r = *(const v16bf*)&tile[wpix][cseg * 16];
    *(v16bf*)(Xw + (size_t)t * DM_ + c0 + cseg * 16) = r;
  }
}

__global__ void cvt_bf16_kernel(const float* __restrict__ s, bf16* __restrict__ d,
                                int n) {
  for (int i = blockIdx.x * blockDim.x + threadIdx.x; i < n;
       i += gridDim.x * blockDim.x)
    d[i] = (bf16)s[i];
}

__global__ void bias_tab_kernel(const float* __restrict__ rpb,
                                const int* __restrict__ rel,
                                float* __restrict__ bt) {
  int i = blockIdx.x * blockDim.x + threadIdx.x;   // NH_ * 64 * 64
  if (i < NH_ * L_ * L_) {
    int n  = i >> 12;
    int lm = i & 4095;
    bt[i] = rpb[n * 225 + rel[lm]];
  }
}

// ---------------- Stage 1: fused QKV projection + attention per (window, head)
__global__ __launch_bounds__(128) void win_attn_kernel(
    const bf16* __restrict__ Xw,
    const bf16* __restrict__ Wqb, const bf16* __restrict__ Wkb,
    const bf16* __restrict__ Wvb,
    const float* __restrict__ bias_t,
    bf16* __restrict__ Obuf) {
  __shared__ bf16  Qs[L_ * HD_] __attribute__((aligned(64)));
  __shared__ bf16  Ks[L_ * HD_] __attribute__((aligned(64)));
  __shared__ bf16  Vts[HD_ * L_] __attribute__((aligned(64)));   // transposed V
  __shared__ float Ss[L_ * L_]  __attribute__((aligned(64)));
  __shared__ bf16  Ps[L_ * L_]  __attribute__((aligned(64)));    // P, then O tile

  const int win  = blockIdx.x;
  const int head = blockIdx.y;
  const int lane = threadIdx.x & 31;
  const int wv   = threadIdx.x >> 5;          // 4 waves, one 16-row M-strip each
  const int drow = 8 * (lane >> 4);
  const int dcol = lane & 15;

  const v8f vzero = {0.f, 0.f, 0.f, 0.f, 0.f, 0.f, 0.f, 0.f};

  // ---- QKV projection: (64 x 768) . W^T slice (64 x 768) per head ----
  const bf16* Wmats[3] = { Wqb + (size_t)head * HD_ * DM_,
                           Wkb + (size_t)head * HD_ * DM_,
                           Wvb + (size_t)head * HD_ * DM_ };
  v8f acc[12];
#pragma unroll
  for (int i = 0; i < 12; ++i) acc[i] = vzero;

  const int row0 = win * L_ + wv * 16;
  for (int kk = 0; kk < DM_; kk += 32) {
    v16bf a = load_A_rm(Xw, DM_, row0, kk, lane);
#pragma unroll
    for (int q = 0; q < 3; ++q) {
#pragma unroll
      for (int nt = 0; nt < 4; ++nt) {
        v16bf b = load_B_nm(Wmats[q], DM_, nt * 16, kk, lane);
        acc[q * 4 + nt] = wmma_bf16(a, b, acc[q * 4 + nt]);
      }
    }
  }
  // scatter to LDS: Q,K row-major (token, d); V transposed (d, token)
#pragma unroll
  for (int nt = 0; nt < 4; ++nt) {
#pragma unroll
    for (int i = 0; i < 8; ++i) {
      int r  = wv * 16 + drow + i;
      int cq = nt * 16 + dcol;
      Qs[r * HD_ + cq]  = (bf16)acc[0 * 4 + nt][i];
      Ks[r * HD_ + cq]  = (bf16)acc[1 * 4 + nt][i];
      Vts[cq * L_ + r]  = (bf16)acc[2 * 4 + nt][i];
    }
  }
  __syncthreads();

  // ---- S = Q . K^T / 8 + bias ----
  v8f sacc[4];
#pragma unroll
  for (int i = 0; i < 4; ++i) sacc[i] = vzero;
  for (int kk = 0; kk < HD_; kk += 32) {
    v16bf a = load_A_rm(Qs, HD_, wv * 16, kk, lane);
#pragma unroll
    for (int nt = 0; nt < 4; ++nt) {
      v16bf b = load_B_nm(Ks, HD_, nt * 16, kk, lane);
      sacc[nt] = wmma_bf16(a, b, sacc[nt]);
    }
  }
  const float* bt = bias_t + (size_t)head * L_ * L_;
#pragma unroll
  for (int nt = 0; nt < 4; ++nt) {
#pragma unroll
    for (int i = 0; i < 8; ++i) {
      int r  = wv * 16 + drow + i;
      int cc = nt * 16 + dcol;
      Ss[r * L_ + cc] = sacc[nt][i] * 0.125f + bt[r * L_ + cc];
    }
  }
  __syncthreads();

  // ---- softmax over each row (64 rows, threads 0..63) ----
  if (threadIdx.x < L_) {
    float* row = Ss + threadIdx.x * L_;
    float mx = -3.4e38f;
    for (int m = 0; m < L_; ++m) mx = fmaxf(mx, row[m]);
    float sum = 0.f;
    for (int m = 0; m < L_; ++m) { float e = __expf(row[m] - mx); row[m] = e; sum += e; }
    float inv = 1.0f / sum;
    for (int m = 0; m < L_; ++m) Ps[threadIdx.x * L_ + m] = (bf16)(row[m] * inv);
  }
  __syncthreads();

  // ---- O = P . V  (B operand from transposed V => contiguous LDS reads) ----
  v8f oacc[4];
#pragma unroll
  for (int i = 0; i < 4; ++i) oacc[i] = vzero;
  for (int kk = 0; kk < L_; kk += 32) {
    v16bf a = load_A_rm(Ps, L_, wv * 16, kk, lane);
#pragma unroll
    for (int nt = 0; nt < 4; ++nt) {
      v16bf b = load_B_nm(Vts, L_, nt * 16, kk, lane);
      oacc[nt] = wmma_bf16(a, b, oacc[nt]);
    }
  }
  __syncthreads();                            // everyone done reading Ps
  // stage O tile in LDS (reuse Ps), then write coalesced 64B chunks
#pragma unroll
  for (int nt = 0; nt < 4; ++nt) {
#pragma unroll
    for (int i = 0; i < 8; ++i)
      Ps[(wv * 16 + drow + i) * L_ + nt * 16 + dcol] = (bf16)oacc[nt][i];
  }
  __syncthreads();
  {
    const int tt  = threadIdx.x >> 1;         // token 0..63
    const int seg = threadIdx.x & 1;          // 32-elem half row
    v16bf a0 = *(const v16bf*)&Ps[tt * L_ + seg * 32];
    v16bf a1 = *(const v16bf*)&Ps[tt * L_ + seg * 32 + 16];
    bf16* dst = Obuf + (size_t)(win * L_ + tt) * DM_ + head * HD_ + seg * 32;
    *(v16bf*)dst        = a0;
    *(v16bf*)(dst + 16) = a1;
  }
}

// ---------------- Stage 2: out = Obuf . Wo^T + b, scattered to (B,C,H,W) -----
__global__ __launch_bounds__(128) void out_proj_kernel(
    const bf16* __restrict__ Obuf, const bf16* __restrict__ Wob,
    const float* __restrict__ bias, float* __restrict__ out) {
  const int win  = blockIdx.x;     // 0..511
  const int nch  = blockIdx.y;     // 0..11 : 64-wide output-channel chunk
  const int lane = threadIdx.x & 31;
  const int wv   = threadIdx.x >> 5;
  const int drow = 8 * (lane >> 4);
  const int dcol = lane & 15;

  const v8f vzero = {0.f, 0.f, 0.f, 0.f, 0.f, 0.f, 0.f, 0.f};
  v8f acc[4];
#pragma unroll
  for (int i = 0; i < 4; ++i) acc[i] = vzero;

  const int row0 = win * L_ + wv * 16;
  for (int kk = 0; kk < DM_; kk += 32) {
    v16bf a = load_A_rm(Obuf, DM_, row0, kk, lane);
#pragma unroll
    for (int nt = 0; nt < 4; ++nt) {
      v16bf b = load_B_nm(Wob, DM_, nch * 64 + nt * 16, kk, lane);
      acc[nt] = wmma_bf16(a, b, acc[nt]);
    }
  }

  const int b  = win >> 6;
  const int wh = (win >> 3) & 7;
  const int ww = win & 7;
  const int tloc0 = wv * 16 + drow;           // multiple of 8: one window row
  const int h = wh * 8 + (tloc0 >> 3);
#pragma unroll
  for (int nt = 0; nt < 4; ++nt) {
    int o = nch * 64 + nt * 16 + dcol;
    float bo = bias[o];
    float* p = out + (((size_t)b * DM_ + o) * 64 + h) * 64 + ww * 8;
    v4f lo = { acc[nt][0] + bo, acc[nt][1] + bo, acc[nt][2] + bo, acc[nt][3] + bo };
    v4f hi = { acc[nt][4] + bo, acc[nt][5] + bo, acc[nt][6] + bo, acc[nt][7] + bo };
    *(v4f*)p       = lo;
    *((v4f*)p + 1) = hi;
  }
}

// ---------------- host-side launch ------------------------------------------
extern "C" void kernel_launch(void* const* d_in, const int* in_sizes, int n_in,
                              void* d_out, int out_size, void* d_ws, size_t ws_size,
                              hipStream_t stream) {
  const float* x     = (const float*)d_in[0];
  const float* Wq    = (const float*)d_in[1];
  const float* Wk    = (const float*)d_in[2];
  const float* Wv    = (const float*)d_in[3];
  const float* Wo_w  = (const float*)d_in[4];
  const float* Wo_b  = (const float*)d_in[5];
  const float* rpb   = (const float*)d_in[6];
  const int*   rel   = (const int*)d_in[7];
  float*       out   = (float*)d_out;

  char* ws = (char*)d_ws;
  const size_t szXw = (size_t)TTOK * DM_ * sizeof(bf16);     // 50.3 MB
  const size_t szW  = (size_t)DM_ * DM_ * sizeof(bf16);      // 1.18 MB each
  bf16*  Xw     = (bf16*)ws;
  bf16*  Wqb    = (bf16*)(ws + szXw);
  bf16*  Wkb    = (bf16*)(ws + szXw + 1 * szW);
  bf16*  Wvb    = (bf16*)(ws + szXw + 2 * szW);
  bf16*  Wob    = (bf16*)(ws + szXw + 3 * szW);
  float* bias_t = (float*)(ws + szXw + 4 * szW);
  bf16*  Obuf   = (bf16*)(ws + szXw + 4 * szW + (size_t)NH_ * L_ * L_ * sizeof(float));

  pack_x_kernel<<<dim3(DM_ / 64, 64, 8), 256, 0, stream>>>(x, Xw);

  const int nW = DM_ * DM_;
  cvt_bf16_kernel<<<(nW + 255) / 256, 256, 0, stream>>>(Wq,   Wqb, nW);
  cvt_bf16_kernel<<<(nW + 255) / 256, 256, 0, stream>>>(Wk,   Wkb, nW);
  cvt_bf16_kernel<<<(nW + 255) / 256, 256, 0, stream>>>(Wv,   Wvb, nW);
  cvt_bf16_kernel<<<(nW + 255) / 256, 256, 0, stream>>>(Wo_w, Wob, nW);

  bias_tab_kernel<<<(NH_ * L_ * L_ + 255) / 256, 256, 0, stream>>>(rpb, rel, bias_t);

  win_attn_kernel<<<dim3(NWIN, NH_), 128, 0, stream>>>(Xw, Wqb, Wkb, Wvb, bias_t, Obuf);

  out_proj_kernel<<<dim3(NWIN, NH_), 128, 0, stream>>>(Obuf, Wob, Wo_b, out);
}
